// ContextualizedNN_73899207295459
// MI455X (gfx1250) — matrix operand order repaired and checked by reading, not compiled, not run on hardware
//
#include <hip/hip_runtime.h>
#include <math.h>

typedef __attribute__((ext_vector_type(16))) _Float16 v16h;
typedef __attribute__((ext_vector_type(8)))  float    v8f;

#define NB 512   // batch
#define LL 50    // padded list length
#define KK 20    // top_k
#define HH 32    // hidden
#define DD 32    // output dim

// One block per batch element b. 5 waves per block; wave w owns channel c=w.
// Each wave loops over (target item + user items), computes the per-item
// channel representation with f16 WMMA (16x16x32: K==H==32), pairs the two
// sub-channels, and accumulates the masked sum in LDS. Final phase does the
// Wout contraction + sigmoid with a 32-lane shuffle reduction.
__global__ __launch_bounds__(160)
void ctxnn_kernel(const float* __restrict__ ctx,
                  const float* __restrict__ W1,
                  const float* __restrict__ b1,
                  const float* __restrict__ W2,
                  const float* __restrict__ b2,
                  const float* __restrict__ Wout,
                  const float* __restrict__ bout,
                  const int*   __restrict__ item_idxs,
                  const int*   __restrict__ user_items,
                  const int*   __restrict__ user_lens,
                  float*       __restrict__ out)
{
    __shared__ float s_acc[100 * 32];     // sum_l u_rep            [n][d]
    __shared__ float s_ir [100 * 32];     // target item rep        [n][d]
    __shared__ float s_o  [5][40 * 32];   // per-channel o staging  [row][d]

    const int tid  = threadIdx.x;
    const int lane = tid & 31;
    const int c    = tid >> 5;      // wave id == channel 0..4
    const int b    = blockIdx.x;
    const int hi   = lane >> 4;     // lane half
    const int lo   = lane & 15;

    for (int i = tid; i < 100 * 32; i += 160) s_acc[i] = 0.0f;
    __syncthreads();

    // Loop-invariant fragments.
    // A-frag (16x32 f16): lane-half hi, element j -> K = (j<8 ? j : j+8) + 8*hi
    // B-frag (32x16 f16): lane-half hi, element j -> Krow = (j<8 ? j : j+8) + 8*hi, Ncol = lo (+16 for tile 1)
    float a1w[16], a1b[16];
    v16h B0, B1;
    #pragma unroll
    for (int j = 0; j < 16; ++j) {
        const int akk  = (j < 8 ? j : j + 8) + 8 * hi;
        a1w[j] = W1[c * HH + akk];
        a1b[j] = b1[c * HH + akk];
        const int krow = (j < 8 ? j : j + 8) + 8 * hi;
        B0[j] = (_Float16)W2[(c * HH + krow) * DD + lo];
        B1[j] = (_Float16)W2[(c * HH + krow) * DD + 16 + lo];
    }
    // Fold the second-layer bias into the WMMA C operand: C/D column is
    // n = lane&15 (+16 for tile 1) for every row held by this lane, so a
    // C fragment with all 8 elements equal to b2[c,n] adds the bias for free.
    const float bb2_0 = b2[c * DD + lo];
    const float bb2_1 = b2[c * DD + 16 + lo];
    v8f C0, C1;
    #pragma unroll
    for (int i = 0; i < 8; ++i) { C0[i] = bb2_0; C1[i] = bb2_1; }

    const int len = user_lens[b];
    const int nit = len + 1;              // it==0 -> target item; else user item it-1

    for (int it = 0; it < nit; ++it) {
        const int item = (it == 0) ? item_idxs[b] : user_items[b * LL + (it - 1)];
        // rows r = p*20 + k  (p in {0,1}, k in [0,20)) -> ctx offset c*40 + r
        const float* base = ctx + (long)item * 200 + c * 40;

        #pragma unroll
        for (int T = 0; T < 3; ++T) {
            const int r  = 16 * T + lo;                 // A row for this lane
            const int ri = (r < 40) ? r : 0;            // branchless: pad rows are
            const float x = base[ri];                   // don't-cares, never read back
            v16h A;
            #pragma unroll
            for (int j = 0; j < 16; ++j)
                A[j] = (_Float16)fmaxf(fmaf(x, a1w[j], a1b[j]), 0.0f);
            v8f D0 = __builtin_amdgcn_wmma_f32_16x16x32_f16(false, A, false, B0, (short)0, C0, false, false);
            v8f D1 = __builtin_amdgcn_wmma_f32_16x16x32_f16(false, A, false, B1, (short)0, C1, false, false);
            // D layout: VGPR i -> row 16T + i + 8*hi, col lo (tile0) / 16+lo (tile1)
            #pragma unroll
            for (int i = 0; i < 8; ++i) {
                const int row = 16 * T + i + 8 * hi;
                if (row < 40) {
                    s_o[c][row * 32 + lo]      = D0[i];
                    s_o[c][row * 32 + 16 + lo] = D1[i];
                }
            }
        }
        __builtin_amdgcn_wave_barrier();   // wave-private LDS staging: order DS ops
        // rep[c,k,d] = o[p=0,k,d] * o[p=1,k,d]; lane owns column d = lane
        for (int k = 0; k < KK; ++k) {
            const float v = s_o[c][k * 32 + lane] * s_o[c][(KK + k) * 32 + lane];
            const int n = c * KK + k;
            if (it == 0) s_ir [n * 32 + lane]  = v;
            else         s_acc[n * 32 + lane] += v;
        }
        __builtin_amdgcn_wave_barrier();
    }

    __syncthreads();

    // pred[b,n] = sigmoid( (sum_d acc[n,d]*ir[n,d]*Wout[d]) / len + bout )
    const float wlane   = Wout[lane];
    const float bo      = bout[0];
    const float inv_len = 1.0f / (float)len;
    for (int n = c; n < 100; n += 5) {
        float v = s_acc[n * 32 + lane] * s_ir[n * 32 + lane] * wlane;
        #pragma unroll
        for (int off = 16; off > 0; off >>= 1)
            v += __shfl_xor(v, off, 32);
        if (lane == 0)
            out[b * 100 + n] = 1.0f / (1.0f + __expf(-(v * inv_len + bo)));
    }
}

extern "C" void kernel_launch(void* const* d_in, const int* in_sizes, int n_in,
                              void* d_out, int out_size, void* d_ws, size_t ws_size,
                              hipStream_t stream) {
    (void)in_sizes; (void)n_in; (void)d_ws; (void)ws_size; (void)out_size;
    const float* ctx  = (const float*)d_in[0];
    const float* W1   = (const float*)d_in[1];
    const float* b1   = (const float*)d_in[2];
    const float* W2   = (const float*)d_in[3];
    const float* b2   = (const float*)d_in[4];
    const float* Wout = (const float*)d_in[5];
    const float* bout = (const float*)d_in[6];
    const int* item_idxs  = (const int*)d_in[7];
    const int* user_items = (const int*)d_in[8];
    const int* user_lens  = (const int*)d_in[9];
    float* out = (float*)d_out;

    ctxnn_kernel<<<NB, 160, 0, stream>>>(ctx, W1, b1, W2, b2, Wout, bout,
                                         item_idxs, user_items, user_lens, out);
}
